// Attention_49838800503089
// MI455X (gfx1250) — matrix operand order repaired and checked by reading, not compiled, not run on hardware
//
#include <hip/hip_runtime.h>
#include <hip/hip_bf16.h>

typedef __attribute__((ext_vector_type(16))) _Float16 v16h;
typedef __attribute__((ext_vector_type(8)))  _Float16 v8h;
typedef __attribute__((ext_vector_type(4)))  _Float16 v4h;
typedef __attribute__((ext_vector_type(8)))  float    v8f;

#define B_    2
#define N_    4096
#define DIM_  768
#define H_    12
#define D_    64
#define BLK_  64
#define NBLK_ 64            // N_/BLK_
#define R_    3
#define NEGV  (-10000.0f)

__device__ __forceinline__ v8f wmma_f16(v16h a, v16h b, v8f c) {
    // 8 args: (neg_a, A, neg_b, B, c_mod, C, reuse_a, reuse_b)
    return __builtin_amdgcn_wmma_f32_16x16x32_f16(false, a, false, b, (short)0, c,
                                                  false, false);
}

// A-fragment (16x32 f16): lane row = lane&15; hi = lane>>4.
// Elements e=0..15 -> K = kbase + 16*hi + e.
// Two contiguous 8-half runs -> two 128-bit loads.
__device__ __forceinline__ v16h load_a16(const _Float16* __restrict__ row,
                                         int kbase, int hi) {
    const v8h lo = *(const v8h*)(row + kbase + 16 * hi);
    const v8h hh = *(const v8h*)(row + kbase + 16 * hi + 8);
    return __builtin_shufflevector(lo, hh, 0, 1, 2, 3, 4, 5, 6, 7,
                                           8, 9, 10, 11, 12, 13, 14, 15);
}

// ---------------------------------------------------------------------------
// Pre-pass 1: X f32 -> f16 (coalesced, 4 elems/thread)
// ---------------------------------------------------------------------------
__global__ __launch_bounds__(256)
void cvt_x_kernel(const float* __restrict__ X, _Float16* __restrict__ X16)
{
    const int i = blockIdx.x * 256 + threadIdx.x;          // 4 elems each
    const float4 v = ((const float4*)X)[i];
    v4h o;
    o[0] = (_Float16)v.x; o[1] = (_Float16)v.y;
    o[2] = (_Float16)v.z; o[3] = (_Float16)v.w;
    ((v4h*)X16)[i] = o;
}

// ---------------------------------------------------------------------------
// Pre-pass 2: W[k][col] f32 -> Wt[col][k] f16 via LDS 32x33 tile transpose
// ---------------------------------------------------------------------------
__global__ __launch_bounds__(256)
void cvt_wt_kernel(const float* __restrict__ W, _Float16* __restrict__ Wt)
{
    __shared__ float tile[32][33];
    const int tx = threadIdx.x & 31;
    const int ty = threadIdx.x >> 5;       // 0..7
    const int k0 = blockIdx.x * 32;
    const int c0 = blockIdx.y * 32;
    #pragma unroll
    for (int r = ty; r < 32; r += 8)
        tile[r][tx] = W[(size_t)(k0 + r) * DIM_ + c0 + tx];
    __syncthreads();
    #pragma unroll
    for (int r = ty; r < 32; r += 8)
        Wt[(size_t)(c0 + r) * DIM_ + k0 + tx] = (_Float16)tile[tx][r];
}

// ---------------------------------------------------------------------------
// Double-buffered GEMM core: 32(M) x 64(N) tile per wave, K = DIM_ in steps
// of 32. Next iteration's fragments are loaded before this iteration's WMMAs
// so the scheduler can overlap 6 outstanding b128-pair loads with math.
// ---------------------------------------------------------------------------
__device__ __forceinline__ void gemm_core_32x64(
    const _Float16* __restrict__ arow0,
    const _Float16* __restrict__ arow1,
    const _Float16* __restrict__ Wt,
    int n0, int col16, int hi,
    v8f acc[2][4])
{
    const _Float16* wcol[4];
    #pragma unroll
    for (int t = 0; t < 4; ++t)
        wcol[t] = Wt + (size_t)(n0 + t * 16 + col16) * DIM_ + 16 * hi;

    v16h a0 = load_a16(arow0, 0, hi);
    v16h a1 = load_a16(arow1, 0, hi);
    v16h b0 = *(const v16h*)(wcol[0]);
    v16h b1 = *(const v16h*)(wcol[1]);
    v16h b2 = *(const v16h*)(wcol[2]);
    v16h b3 = *(const v16h*)(wcol[3]);

    for (int kk = 0; kk < DIM_; kk += 32) {
        const int kn = (kk + 32 < DIM_) ? kk + 32 : 0;   // wrap: harmless
        const v16h na0 = load_a16(arow0, kn, hi);
        const v16h na1 = load_a16(arow1, kn, hi);
        const v16h nb0 = *(const v16h*)(wcol[0] + kn);
        const v16h nb1 = *(const v16h*)(wcol[1] + kn);
        const v16h nb2 = *(const v16h*)(wcol[2] + kn);
        const v16h nb3 = *(const v16h*)(wcol[3] + kn);

        acc[0][0] = wmma_f16(a0, b0, acc[0][0]);
        acc[1][0] = wmma_f16(a1, b0, acc[1][0]);
        acc[0][1] = wmma_f16(a0, b1, acc[0][1]);
        acc[1][1] = wmma_f16(a1, b1, acc[1][1]);
        acc[0][2] = wmma_f16(a0, b2, acc[0][2]);
        acc[1][2] = wmma_f16(a1, b2, acc[1][2]);
        acc[0][3] = wmma_f16(a0, b3, acc[0][3]);
        acc[1][3] = wmma_f16(a1, b3, acc[1][3]);

        a0 = na0; a1 = na1;
        b0 = nb0; b1 = nb1; b2 = nb2; b3 = nb3;
    }
}

// ---------------------------------------------------------------------------
// Kernel 1: projection GEMM.
//   out = cvt_f16( (A16 @ Wt^T + bias) * scale )
// vmode 0: head-major [B][H][N][D] (Q, K)   vmode 1: transposed [B][H][D][N] (V)
// ---------------------------------------------------------------------------
__global__ __launch_bounds__(32)
void qkv_gemm_kernel(const _Float16* __restrict__ A16,  // [B*N][DIM]
                     const _Float16* __restrict__ Wt,   // [DIM(col)][DIM(k)]
                     const float* __restrict__ bias,
                     _Float16* __restrict__ out,
                     float scale, int vmode)
{
    const int lane  = threadIdx.x & 31;
    const int col16 = lane & 15;
    const int hi    = lane >> 4;
    const int m0    = blockIdx.x * 32;
    const int n0    = blockIdx.y * 64;

    v8f acc[2][4] = {};
    const _Float16* arow0 = A16 + (size_t)(m0 + col16) * DIM_;
    const _Float16* arow1 = arow0 + (size_t)16 * DIM_;

    gemm_core_32x64(arow0, arow1, Wt, n0, col16, hi, acc);

    #pragma unroll
    for (int g = 0; g < 2; ++g) {
        #pragma unroll
        for (int t = 0; t < 4; ++t) {
            const int col = n0 + t * 16 + col16;
            const int h   = col >> 6;
            const int d   = col & 63;
            const float bs = bias[col];
            #pragma unroll
            for (int i = 0; i < 8; ++i) {
                const int m   = m0 + g * 16 + i + 8 * hi;
                const int bb  = m >> 12;
                const int tok = m & 4095;
                const float v = (acc[g][t][i] + bs) * scale;
                const size_t idx = vmode
                    ? (((size_t)bb * H_ + h) * D_ + d) * N_ + tok   // V^T
                    : (((size_t)bb * H_ + h) * N_ + tok) * D_ + d;  // Q,K
                out[idx] = (_Float16)v;
            }
        }
    }
}

// ---------------------------------------------------------------------------
// Kernel 2: BigBird block-sparse flash attention.
// Grid: (qblock, head, batch); 128 threads = 4 waves; wave w owns query rows
// [w*16, w*16+16). K fragments and Vt fragments are single 32B loads.
// ---------------------------------------------------------------------------
__global__ __launch_bounds__(128)
void bigbird_attn_kernel(const _Float16* __restrict__ Q,   // [B][H][N][D]
                         const _Float16* __restrict__ K,   // [B][H][N][D]
                         const _Float16* __restrict__ Vt,  // [B][H][D][N]
                         const float* __restrict__ mask,
                         const int* __restrict__ rand_attn,
                         _Float16* __restrict__ Aout)      // [B][N][H*D]
{
    __shared__ float s_lds[4][16][68];                        // padded
    __shared__ __attribute__((aligned(32))) _Float16 p_lds[4][16][72];
    __shared__ float corr_lds[4][16];
    __shared__ float lsum_lds[4][16];

    const int qi   = blockIdx.x;
    const int h    = blockIdx.y;
    const int b    = blockIdx.z;
    const int wave = threadIdx.x >> 5;
    const int lane = threadIdx.x & 31;
    const int col16 = lane & 15;
    const int hi    = lane >> 4;

    const size_t headbase = ((size_t)b * H_ + h) * (size_t)N_ * D_;
    const _Float16* Qh  = Q  + headbase;
    const _Float16* Kh  = K  + headbase;
    const _Float16* Vth = Vt + headbase;   // [D][N]

    const int qrow0 = qi * BLK_ + wave * 16;

    // Q A-fragments (two K-chunks over D=64)
    const _Float16* qrow = Qh + (size_t)(qrow0 + col16) * D_;
    v16h qa[2];
    #pragma unroll
    for (int c = 0; c < 2; ++c)
        qa[c] = load_a16(qrow, c * 32, hi);

    // Key-block list (duplicates kept to match reference concatenation).
    int list[8] = {0};
    int nkb;
    const bool dense = (qi == 0) || (qi == NBLK_ - 1);
    if (dense) {
        nkb = NBLK_;
    } else {
        const int rbase = (h * (NBLK_ - 2) + (qi - 1)) * R_;
        const int r0 = rand_attn[rbase + 0];
        const int r1 = rand_attn[rbase + 1];
        const int r2 = rand_attn[rbase + 2];
        if (qi == 1) {
            list[0]=0; list[1]=1; list[2]=2; list[3]=NBLK_-1;
            list[4]=r0; list[5]=r1; list[6]=r2; nkb = 7;
        } else if (qi == NBLK_ - 2) {
            list[0]=0; list[1]=NBLK_-3; list[2]=NBLK_-2; list[3]=NBLK_-1;
            list[4]=r0; list[5]=r1; list[6]=r2; nkb = 7;
        } else {
            list[0]=0; list[1]=qi-1; list[2]=qi; list[3]=qi+1;
            list[4]=r0; list[5]=r1; list[6]=r2; list[7]=NBLK_-1; nkb = 8;
        }
    }

    float mrun = -1e30f;   // valid in lanes 0..15
    float lrun = 0.0f;
    v8f oacc[4] = {};

    for (int it = 0; it < nkb; ++it) {
        const int kb  = dense ? it : list[it];
        const int kt0 = kb * BLK_;

        if (it + 1 < nkb) {
            const int nb = dense ? (it + 1) : list[it + 1];
            __builtin_prefetch(Kh  + (size_t)(nb * BLK_ + col16) * D_, 0, 1);
            __builtin_prefetch(Vth + (size_t)col16 * N_ + nb * BLK_, 0, 1);
        }

        // ---- S = (Q * rsqrt(D)) @ K^T : 4 key tiles x 2 K-chunks ----
        v8f s[4] = {};
        #pragma unroll
        for (int t = 0; t < 4; ++t) {
            const _Float16* krow = Kh + (size_t)(kt0 + t * 16 + col16) * D_;
            #pragma unroll
            for (int c = 0; c < 2; ++c) {
                const v16h kfr = *(const v16h*)(krow + c * 32 + 16 * hi);
                s[t] = wmma_f16(qa[c], kfr, s[t]);
            }
        }

        // ---- key mask + spill S to LDS (C layout: col=lane&15, row=i+8*hi)
        #pragma unroll
        for (int t = 0; t < 4; ++t) {
            const int kin  = t * 16 + col16;
            const float mk = mask[(size_t)b * N_ + kt0 + kin];
            const float pen = (1.0f - mk) * NEGV;
            #pragma unroll
            for (int i = 0; i < 8; ++i)
                s_lds[wave][i + 8 * hi][kin] = s[t][i] + pen;
        }
        __syncthreads();

        // ---- online softmax, one row per lane (lanes 0..15) ----
        if (lane < 16) {
            const int r = lane;
            float bm = -1e30f;
            for (int j = 0; j < 64; ++j)
                bm = fmaxf(bm, s_lds[wave][r][j]);
            const float mnew = fmaxf(mrun, bm);
            const float corr = __expf(mrun - mnew);
            float psum = 0.0f;
            for (int j = 0; j < 64; ++j) {
                const float p = __expf(s_lds[wave][r][j] - mnew);
                psum += p;
                p_lds[wave][r][j] = (_Float16)p;
            }
            lrun = lrun * corr + psum;
            mrun = mnew;
            corr_lds[wave][r] = corr;
        }
        __syncthreads();

        // ---- rescale O accumulators ----
        #pragma unroll
        for (int i = 0; i < 8; ++i) {
            const float cf = corr_lds[wave][i + 8 * hi];
            #pragma unroll
            for (int t = 0; t < 4; ++t)
                oacc[t][i] = oacc[t][i] * cf;
        }

        // ---- O += P @ V : 2 K-chunks x 4 d tiles ----
        const _Float16* prow = &p_lds[wave][col16][0];
        #pragma unroll
        for (int c = 0; c < 2; ++c) {
            const v16h pfr = load_a16(prow, c * 32, hi);   // ds_load_b128 x2
            #pragma unroll
            for (int t = 0; t < 4; ++t) {
                const _Float16* vrow = Vth + (size_t)(t * 16 + col16) * N_;
                const v16h vfr = *(const v16h*)(vrow + kt0 + c * 32 + 16 * hi);
                oacc[t] = wmma_f16(pfr, vfr, oacc[t]);
            }
        }
        // next iteration's S-store is ordered after this iteration's S reads
        // by the post-softmax barrier; P/corr reads complete before the next
        // post-S-store barrier.
    }

    // ---- normalize, apply query mask, store ctx as f16 [B][N][H*D] ----
    if (lane < 16) {
        const float qm = mask[(size_t)b * N_ + qrow0 + lane];
        lsum_lds[wave][lane] = qm / lrun;
    }
    __syncthreads();

    #pragma unroll
    for (int i = 0; i < 8; ++i) {
        const int r = i + 8 * hi;
        const float f = lsum_lds[wave][r];
        const int tok = qrow0 + r;
        #pragma unroll
        for (int t = 0; t < 4; ++t) {
            const int d = t * 16 + col16;
            Aout[((size_t)b * N_ + tok) * (H_ * D_) + h * D_ + d] =
                (_Float16)(oacc[t][i] * f);
        }
    }
}

// ---------------------------------------------------------------------------
// Kernel 3: output projection  d_out = ctx(f16) @ Wff + bff  (f32 out)
// ---------------------------------------------------------------------------
__global__ __launch_bounds__(32)
void ff_gemm_kernel(const _Float16* __restrict__ A16,   // [B*N][DIM]
                    const _Float16* __restrict__ Wt,    // [DIM(col)][DIM(k)]
                    const float* __restrict__ bias,
                    float* __restrict__ out)            // [B*N][DIM]
{
    const int lane  = threadIdx.x & 31;
    const int col16 = lane & 15;
    const int hi    = lane >> 4;
    const int m0    = blockIdx.x * 32;
    const int n0    = blockIdx.y * 64;

    v8f acc[2][4] = {};
    const _Float16* arow0 = A16 + (size_t)(m0 + col16) * DIM_;
    const _Float16* arow1 = arow0 + (size_t)16 * DIM_;

    gemm_core_32x64(arow0, arow1, Wt, n0, col16, hi, acc);

    #pragma unroll
    for (int g = 0; g < 2; ++g) {
        #pragma unroll
        for (int t = 0; t < 4; ++t) {
            const int col = n0 + t * 16 + col16;
            const float bs = bias[col];
            #pragma unroll
            for (int i = 0; i < 8; ++i) {
                const int m = m0 + g * 16 + i + 8 * hi;
                out[(size_t)m * DIM_ + col] = acc[g][t][i] + bs;
            }
        }
    }
}

// ---------------------------------------------------------------------------
extern "C" void kernel_launch(void* const* d_in, const int* in_sizes, int n_in,
                              void* d_out, int out_size, void* d_ws, size_t ws_size,
                              hipStream_t stream)
{
    const float* X    = (const float*)d_in[0];
    const float* mask = (const float*)d_in[1];
    const int*   rnd  = (const int*)  d_in[2];
    const float* Wq   = (const float*)d_in[3];
    const float* bq   = (const float*)d_in[4];
    const float* Wk   = (const float*)d_in[5];
    const float* bk   = (const float*)d_in[6];
    const float* Wv   = (const float*)d_in[7];
    const float* bv   = (const float*)d_in[8];
    const float* Wff  = (const float*)d_in[9];
    const float* bff  = (const float*)d_in[10];
    float* out = (float*)d_out;

    _Float16* ws = (_Float16*)d_ws;
    const size_t per  = (size_t)B_ * H_ * N_ * D_;   // 6,291,456 halves
    const size_t wsz  = (size_t)DIM_ * DIM_;         // 589,824 halves
    _Float16* q    = ws;
    _Float16* k    = ws + per;
    _Float16* vt   = ws + 2 * per;                   // [B][H][D][N]
    _Float16* ctx  = ws + 3 * per;                   // [B*N][H*D]
    _Float16* x16  = ws + 4 * per;                   // [B*N][DIM]
    _Float16* wqT  = ws + 5 * per;
    _Float16* wkT  = wqT + wsz;
    _Float16* wvT  = wkT + wsz;
    _Float16* wfT  = wvT + wsz;

    // conversions
    cvt_x_kernel<<<(B_ * N_ * DIM_) / (256 * 4), 256, 0, stream>>>(X, x16);
    dim3 gw(DIM_ / 32, DIM_ / 32, 1);                // 24 x 24
    cvt_wt_kernel<<<gw, 256, 0, stream>>>(Wq,  wqT);
    cvt_wt_kernel<<<gw, 256, 0, stream>>>(Wk,  wkT);
    cvt_wt_kernel<<<gw, 256, 0, stream>>>(Wv,  wvT);
    cvt_wt_kernel<<<gw, 256, 0, stream>>>(Wff, wfT);

    // projections (Q pre-scaled by rsqrt(D) = 0.125)
    dim3 gproj((B_ * N_) / 32, DIM_ / 64, 1);        // 256 x 12
    qkv_gemm_kernel<<<gproj, 32, 0, stream>>>(x16, wqT, bq, q,  0.125f, 0);
    qkv_gemm_kernel<<<gproj, 32, 0, stream>>>(x16, wkT, bk, k,  1.0f,   0);
    qkv_gemm_kernel<<<gproj, 32, 0, stream>>>(x16, wvT, bv, vt, 1.0f,   1);

    // sparse attention
    dim3 gattn(NBLK_, H_, B_);                       // 64 x 12 x 2
    bigbird_attn_kernel<<<gattn, 128, 0, stream>>>(q, k, vt, mask, rnd, ctx);

    // output projection
    ff_gemm_kernel<<<gproj, 32, 0, stream>>>(ctx, wfT, bff, out);
}
